// RNNModel_24902220382770
// MI455X (gfx1250) — matrix operand order repaired and checked by reading, not compiled, not run on hardware
//
#include <hip/hip_runtime.h>
#include <hip/hip_bf16.h>

// ---------------------------------------------------------------------------
// Problem constants (from reference)
// ---------------------------------------------------------------------------
#define VV 50257
#define EE 1024
#define HH 1024
#define SS 128
#define BB 16

typedef __bf16 bf16_t;
typedef bf16_t v8bf  __attribute__((ext_vector_type(8)));
typedef bf16_t v16bf __attribute__((ext_vector_type(16)));
typedef float  v8f   __attribute__((ext_vector_type(8)));

// ---------------------------------------------------------------------------
// WMMA fragment loaders (bf16, 16x16x32, wave32)
// ---------------------------------------------------------------------------
__device__ __forceinline__ v16bf concat8(v8bf lo, v8bf hi) {
    return __builtin_shufflevector(lo, hi, 0, 1, 2, 3, 4, 5, 6, 7,
                                           8, 9, 10, 11, 12, 13, 14, 15);
}

// A-matrix fragment: 16x32 tile of row-major A (ld elems), tile origin (row0, k0).
// Per ISA 7.12.2: lanes 0-15 hold M=lane, K {0..7,16..23}; lanes 16-31 K {8..15,24..31}.
__device__ __forceinline__ v16bf load_a_frag(const bf16_t* __restrict__ A,
                                             size_t row0, int ld, int k0, int lane) {
    const bf16_t* p = A + (row0 + (size_t)(lane & 15)) * (size_t)ld + k0 + ((lane >> 4) << 3);
    v8bf lo = *(const v8bf*)p;
    v8bf hi = *(const v8bf*)(p + 16);
    return concat8(lo, hi);
}

// B-matrix fragment: result = A @ W^T, W stored (N,K) row-major -> B[k][n] = W[n][k].
// Per ISA B layout: lanes 0-15 hold N=lane, K {0..7,16..23}; lanes 16-31 hold N=lane-16, K {8..15,24..31}.
// Row index clamped to nmax so ragged-N tiles never read OOB (masked at store).
__device__ __forceinline__ v16bf load_b_frag(const bf16_t* __restrict__ W,
                                             int n0, int nmax, int ld, int k0, int lane) {
    int r = n0 + (lane & 15);
    if (r > nmax) r = nmax;
    const bf16_t* p = W + (size_t)r * (size_t)ld + k0 + ((lane >> 4) << 3);
    v8bf lo = *(const v8bf*)p;
    v8bf hi = *(const v8bf*)(p + 16);
    return concat8(lo, hi);
}

__device__ __forceinline__ float sigf(float x) { return 1.0f / (1.0f + __expf(-x)); }

// ---------------------------------------------------------------------------
// Generic bf16 WMMA GEMM:  out = A(MxK) @ W(NxK)^T  (+bias, custom epilogues)
// Block 256 thr = 8 waves in 2x4 grid; wave tile 64x64 (16 v8f accumulators).
// MODE 0: fp32 out (ld=N) + bias1 + bias2 (nullable)       [proj / decoder]
// MODE 1: trans epilogue: + bias1(bhh2) + mvec[b], t<k mask selects hs, bf16 out
// ---------------------------------------------------------------------------
template <int MODE>
__global__ void __launch_bounds__(256) gemm_bf16_wmma(
    const bf16_t* __restrict__ A, const bf16_t* __restrict__ W,
    const float* __restrict__ bias1, const float* __restrict__ bias2,
    const float* __restrict__ mvec, const float* __restrict__ hsf,
    const int* __restrict__ kptr,
    float* __restrict__ outf, bf16_t* __restrict__ outbf,
    int M, int N, int K)
{
    const int lane = threadIdx.x & 31;
    const int w    = threadIdx.x >> 5;
    const int m0   = blockIdx.y * 128 + (w & 1) * 64;
    const int n0   = blockIdx.x * 256 + (w >> 1) * 64;

    v8f acc[4][4];
    const v8f vz = {0.f, 0.f, 0.f, 0.f, 0.f, 0.f, 0.f, 0.f};
#pragma unroll
    for (int i = 0; i < 4; ++i)
#pragma unroll
        for (int j = 0; j < 4; ++j) acc[i][j] = vz;

    for (int k0 = 0; k0 < K; k0 += 32) {
        v16bf a[4], b[4];
#pragma unroll
        for (int i = 0; i < 4; ++i)
            a[i] = load_a_frag(A, (size_t)(m0 + i * 16), K, k0, lane);
#pragma unroll
        for (int j = 0; j < 4; ++j)
            b[j] = load_b_frag(W, n0 + j * 16, N - 1, K, k0, lane);
#pragma unroll
        for (int i = 0; i < 4; ++i)
#pragma unroll
            for (int j = 0; j < 4; ++j)
                acc[i][j] = __builtin_amdgcn_wmma_f32_16x16x32_bf16(
                    false, a[i], false, b[j], (short)0, acc[i][j], false, false);
    }

    // C/D layout: lanes 0-15 -> (M=e, N=lane); lanes 16-31 -> (M=8+e, N=lane-16)
    const int mb  = (lane >> 4) * 8;
    const int col = lane & 15;
    const int kval = (MODE == 1) ? kptr[0] : 0;

#pragma unroll
    for (int i = 0; i < 4; ++i)
#pragma unroll
        for (int j = 0; j < 4; ++j)
#pragma unroll
            for (int e = 0; e < 8; ++e) {
                int row = m0 + i * 16 + mb + e;
                int nc  = n0 + j * 16 + col;
                if (nc < N) {
                    float v = acc[i][j][e];
                    if (bias1) v += bias1[nc];
                    if (bias2) v += bias2[nc];
                    if (MODE == 0) {
                        outf[(size_t)row * (size_t)N + nc] = v;
                    } else {
                        int s = row >> 4, bidx = row & 15;
                        v += mvec[bidx * HH + nc];
                        if (s < kval) v = hsf[(size_t)row * HH + nc];
                        outbf[(size_t)row * HH + nc] = (bf16_t)v;
                    }
                }
            }
}

// ---------------------------------------------------------------------------
// LSTM step: gates = G0[t] + h_{t-1} @ W_hh^T; elementwise recurrence.
// 16 blocks x 128 thr; block owns 64 hidden units; wave w computes gate w
// (i,f,g,o) for those units via WMMA over K=1024; combine via LDS.
// ---------------------------------------------------------------------------
__global__ void __launch_bounds__(128) lstm_step_kernel(
    const bf16_t* __restrict__ hprev_bf,  // 16 x 1024 bf16
    const bf16_t* __restrict__ Whh_bf,    // 4096 x 1024 bf16
    const float* __restrict__ G0,         // (S*B) x 4096 fp32 (biases included)
    float* __restrict__ cbuf,             // 16 x 1024 fp32 (in/out)
    float* __restrict__ hs_f,             // (S*B) x H fp32
    bf16_t* __restrict__ hs_bf,           // (S*B) x H bf16
    int t)
{
    __shared__ float gbuf[4][16][64];

    const int lane = threadIdx.x & 31;
    const int w    = threadIdx.x >> 5;   // gate index: 0=i,1=f,2=g,3=o
    const int u0   = blockIdx.x * 64;    // hidden-unit base

    v8f acc[4];
    const v8f vz = {0.f, 0.f, 0.f, 0.f, 0.f, 0.f, 0.f, 0.f};
#pragma unroll
    for (int j = 0; j < 4; ++j) acc[j] = vz;

    for (int k0 = 0; k0 < HH; k0 += 32) {
        v16bf a = load_a_frag(hprev_bf, 0, HH, k0, lane);
#pragma unroll
        for (int j = 0; j < 4; ++j) {
            v16bf b = load_b_frag(Whh_bf, w * HH + u0 + j * 16, 4 * HH - 1, HH, k0, lane);
            acc[j] = __builtin_amdgcn_wmma_f32_16x16x32_bf16(
                false, a, false, b, (short)0, acc[j], false, false);
        }
    }

    const int mb  = (lane >> 4) * 8;
    const int col = lane & 15;
#pragma unroll
    for (int j = 0; j < 4; ++j)
#pragma unroll
        for (int e = 0; e < 8; ++e) {
            int m = mb + e;
            int n = u0 + j * 16 + col;                  // unit index within H
            float v = acc[j][e] + G0[(size_t)(t * BB + m) * (4 * HH) + w * HH + n];
            gbuf[w][m][j * 16 + col] = v;
        }
    __syncthreads();

    for (int idx = threadIdx.x; idx < 16 * 64; idx += 128) {
        int m = idx >> 6, u = idx & 63;
        float gi = gbuf[0][m][u], gf = gbuf[1][m][u];
        float gg = gbuf[2][m][u], go = gbuf[3][m][u];
        float c  = cbuf[m * HH + u0 + u];
        float cn = sigf(gf) * c + sigf(gi) * tanhf(gg);
        float hn = sigf(go) * tanhf(cn);
        cbuf[m * HH + u0 + u] = cn;
        size_t ho = (size_t)(t * BB + m) * HH + u0 + u;
        hs_f[ho]  = hn;
        hs_bf[ho] = (bf16_t)hn;
    }
}

// ---------------------------------------------------------------------------
// Small helper kernels
// ---------------------------------------------------------------------------
__global__ void f32_to_bf16_kernel(const float* __restrict__ src,
                                   bf16_t* __restrict__ dst, int n) {
    int i = blockIdx.x * blockDim.x + threadIdx.x;
    if (i < n) dst[i] = (bf16_t)src[i];
}

__global__ void embed_kernel(const int* __restrict__ tokens,
                             const float* __restrict__ table,
                             bf16_t* __restrict__ emb_bf) {
    int row = blockIdx.x;                       // 0..S*B-1 (s*B+b)
    int tok = tokens[row];
    const float* src = table + (size_t)tok * EE;
    for (int e = threadIdx.x; e < EE; e += blockDim.x)
        emb_bf[(size_t)row * EE + e] = (bf16_t)src[e];
}

__global__ void init_state_kernel(const float* __restrict__ h0,
                                  const float* __restrict__ c0,
                                  bf16_t* __restrict__ h0bf,
                                  float* __restrict__ cbuf, int n) {
    int i = blockIdx.x * blockDim.x + threadIdx.x;
    if (i < n) { h0bf[i] = (bf16_t)h0[i]; cbuf[i] = c0[i]; }
}

// m = h0 @ Whm^T + bhm   (16 x 1024, exact fp32; tiny)
__global__ void h0whm_kernel(const float* __restrict__ h0,
                             const float* __restrict__ Whm,
                             const float* __restrict__ bhm,
                             float* __restrict__ mvec) {
    int idx = blockIdx.x * blockDim.x + threadIdx.x;   // 16384
    int b = idx >> 10, n = idx & 1023;
    float s = bhm[n];
    const float* hr = h0 + (size_t)b * HH;
    const float* wr = Whm + (size_t)n * HH;
    for (int k = 0; k < HH; ++k) s += hr[k] * wr[k];
    mvec[idx] = s;
}

// ---------------------------------------------------------------------------
// Host launcher
// ---------------------------------------------------------------------------
extern "C" void kernel_launch(void* const* d_in, const int* in_sizes, int n_in,
                              void* d_out, int out_size, void* d_ws, size_t ws_size,
                              hipStream_t stream) {
    (void)in_sizes; (void)n_in; (void)out_size; (void)ws_size;

    const int*   tokens    = (const int*)d_in[0];
    const float* h0        = (const float*)d_in[1];
    const float* c0        = (const float*)d_in[2];
    const float* emb_table = (const float*)d_in[3];
    const float* W_ih      = (const float*)d_in[4];
    const float* W_hh      = (const float*)d_in[5];
    const float* b_ih      = (const float*)d_in[6];
    const float* b_hh      = (const float*)d_in[7];
    const float* Whh2      = (const float*)d_in[8];
    const float* bhh2      = (const float*)d_in[9];
    const float* Whm       = (const float*)d_in[10];
    const float* bhm       = (const float*)d_in[11];
    const float* decW      = (const float*)d_in[12];
    const float* decb      = (const float*)d_in[13];
    const int*   kptr      = (const int*)d_in[14];
    float*       out       = (float*)d_out;

    char* ws = (char*)d_ws;
    // Persistent through decoder:
    bf16_t* OUTBF  = (bf16_t*)(ws + 0);                       //  4,194,304 B
    // Phase-1 region (union with decW_bf16, which is only needed after trans):
    const size_t P1 = 4u * 1024u * 1024u;
    bf16_t* WIHBF  = (bf16_t*)(ws + P1 + 0);                  //  8,388,608 B
    bf16_t* WHHBF  = (bf16_t*)(ws + P1 + 8388608);            //  8,388,608 B
    bf16_t* EMBBF  = (bf16_t*)(ws + P1 + 16777216);           //  4,194,304 B
    float*  G0     = (float*) (ws + P1 + 20971520);           // 33,554,432 B
    float*  HSF    = (float*) (ws + P1 + 54525952);           //  8,388,608 B
    bf16_t* HSBF   = (bf16_t*)(ws + P1 + 62914560);           //  4,194,304 B
    bf16_t* WHH2BF = (bf16_t*)(ws + P1 + 67108864);           //  2,097,152 B
    float*  MVEC   = (float*) (ws + P1 + 69206016);           //     65,536 B
    bf16_t* H0BF   = (bf16_t*)(ws + P1 + 69271552);           //     32,768 B
    float*  CBUF   = (float*) (ws + P1 + 69304320);           //     65,536 B
    bf16_t* DECWBF = (bf16_t*)(ws + P1);                      // 102,926,336 B (reuses phase-1)

    // --- weight conversions (bf16) ---
    {
        int n = 4 * HH * EE;
        f32_to_bf16_kernel<<<(n + 255) / 256, 256, 0, stream>>>(W_ih, WIHBF, n);
        n = 4 * HH * HH;
        f32_to_bf16_kernel<<<(n + 255) / 256, 256, 0, stream>>>(W_hh, WHHBF, n);
        n = HH * HH;
        f32_to_bf16_kernel<<<(n + 255) / 256, 256, 0, stream>>>(Whh2, WHH2BF, n);
    }

    // --- embedding gather -> bf16 ---
    embed_kernel<<<SS * BB, 256, 0, stream>>>(tokens, emb_table, EMBBF);

    // --- initial state ---
    init_state_kernel<<<(BB * HH + 255) / 256, 256, 0, stream>>>(h0, c0, H0BF, CBUF, BB * HH);

    // --- G0 = emb @ W_ih^T + b_ih + b_hh  (2048 x 4096 x 1024) ---
    {
        dim3 g((4 * HH + 255) / 256, (SS * BB + 127) / 128);
        gemm_bf16_wmma<0><<<g, 256, 0, stream>>>(
            EMBBF, WIHBF, b_ih, b_hh, nullptr, nullptr, nullptr,
            G0, nullptr, SS * BB, 4 * HH, EE);
    }

    // --- m = h0 @ Whm^T + bhm ---
    h0whm_kernel<<<(BB * HH + 255) / 256, 256, 0, stream>>>(h0, Whm, bhm, MVEC);

    // --- sequential LSTM ---
    for (int t = 0; t < SS; ++t) {
        const bf16_t* hprev = (t == 0) ? H0BF : (HSBF + (size_t)(t - 1) * BB * HH);
        lstm_step_kernel<<<HH / 64, 128, 0, stream>>>(
            hprev, WHHBF, G0, CBUF, HSF, HSBF, t);
    }

    // --- trans = hs @ Whh2^T + bhh2 + m; mask t<k -> hs; output bf16 ---
    {
        dim3 g((HH + 255) / 256, (SS * BB + 127) / 128);
        gemm_bf16_wmma<1><<<g, 256, 0, stream>>>(
            HSBF, WHH2BF, bhh2, nullptr, MVEC, HSF, kptr,
            nullptr, OUTBF, SS * BB, HH, HH);
    }

    // --- decW -> bf16 (phase-1 buffers dead now) ---
    {
        int n = VV * HH;
        f32_to_bf16_kernel<<<(n + 255) / 256, 256, 0, stream>>>(decW, DECWBF, n);
    }

    // --- decoded = outputs @ decW^T + decb  (2048 x 50257 x 1024) ---
    {
        dim3 g((VV + 255) / 256, (SS * BB + 127) / 128);
        gemm_bf16_wmma<0><<<g, 256, 0, stream>>>(
            OUTBF, DECWBF, decb, nullptr, nullptr, nullptr, nullptr,
            out, nullptr, SS * BB, VV, HH);
    }
}
